// FrameSimilarity_21998822490744
// MI455X (gfx1250) — matrix operand order repaired and checked by reading, not compiled, not run on hardware
//
#include <hip/hip_runtime.h>

// ---------------------------------------------------------------------------
// Types for CDNA5 WMMA (gfx1250, wave32)
// ---------------------------------------------------------------------------
typedef __attribute__((ext_vector_type(16))) __bf16 v16bf;
typedef __attribute__((ext_vector_type(8)))  __bf16 v8bf;
typedef __attribute__((ext_vector_type(8)))  float  v8f;

#define T_DIM   512
#define B_DIM   4
#define C2      2048      // concat channel dim (2*1024)
#define D_DIM   128       // projection dim
#define HW      16
#define LWIN    101
#define PAD     50        // (LWIN-1)/2

__device__ __forceinline__ __bf16 f2bf(float f) {
    union { float f; unsigned u; } v; v.f = f;
    unsigned r = v.u + 0x7fffu + ((v.u >> 16) & 1u);   // round-to-nearest-even
    unsigned short h = (unsigned short)(r >> 16);
    return __builtin_bit_cast(__bf16, h);
}

// ---------------------------------------------------------------------------
// Kernel P: weight precasting.  W_proj (128x2048 f32) -> bf16,
//           W_fc (128x101 f32) -> bf16 zero-padded to (128x128).
// ---------------------------------------------------------------------------
__global__ void prep_kernel(const float* __restrict__ Wp, const float* __restrict__ Wfc,
                            __bf16* __restrict__ WpB, __bf16* __restrict__ WfcB) {
    int gid = blockIdx.x * 256 + threadIdx.x;          // 128*2048 threads
    if (gid < D_DIM * C2) WpB[gid] = f2bf(Wp[gid]);
    if (gid < D_DIM * 128) {
        int o = gid >> 7, w = gid & 127;
        WfcB[gid] = (w < LWIN) ? f2bf(Wfc[o * LWIN + w]) : f2bf(0.0f);
    }
}

// ---------------------------------------------------------------------------
// Kernel A: spatial mean + concat + transpose.
// Each thread reduces one aligned 64B cacheline (16 contiguous f32) and writes
// one bf16 of F[row=b*T+t][c].  Writes are coalesced across lanes.
// Dominant HBM traffic of the whole problem: 268 MB read.
// ---------------------------------------------------------------------------
__global__ void mean_kernel(const float* __restrict__ x1, const float* __restrict__ x2,
                            __bf16* __restrict__ F) {
    int gid = blockIdx.x * 256 + threadIdx.x;          // 2048*2048 threads
    int row = gid >> 11;                               // b*T + t
    int c   = gid & (C2 - 1);
    int b   = row >> 9;
    int t   = row & (T_DIM - 1);
    const float* src = (c < 1024) ? x1 : x2;
    int cc = c & 1023;
    const float4* p = (const float4*)(src + ((size_t)((b * 1024 + cc) * T_DIM + t) << 4));
    float4 s0 = p[0], s1 = p[1], s2 = p[2], s3 = p[3];
    float s = (s0.x + s0.y + s0.z + s0.w) + (s1.x + s1.y + s1.z + s1.w)
            + (s2.x + s2.y + s2.z + s2.w) + (s3.x + s3.y + s3.z + s3.w);
    F[(size_t)row * C2 + c] = f2bf(s * 0.0625f);
}

// ---------------------------------------------------------------------------
// Kernel B: projection GEMM (2048 x 2048) * W_proj^T (2048 x 128) + row L2-norm.
// 128 workgroups (16-row M tiles) x 8 waves (one 16-col N tile each).
// A fragment (16x32 bf16): lane m=l&15, h=l>>4; elems 0..7 -> K=kb+8h+e,
//                          elems 8..15 -> K=kb+16+8h+(e-8).
// B fragment (32x16 bf16): lane col n=l&15; elems 0..7 -> K=kb+8h+e, 8..15 -> K=kb+16+8h+(e-8).
// ---------------------------------------------------------------------------
__global__ void proj_kernel(const __bf16* __restrict__ F, const __bf16* __restrict__ WpB,
                            __bf16* __restrict__ Xn) {
    __shared__ float rowNorm[16];
    int wave = threadIdx.x >> 5;
    int lane = threadIdx.x & 31;
    int m = lane & 15, h = lane >> 4;
    int mtile = blockIdx.x;                            // rows mtile*16 .. +15
    int rowA  = mtile * 16 + m;
    int ncol  = wave * 16 + m;                         // B column = W_proj row

    if (threadIdx.x < 16) rowNorm[threadIdx.x] = 0.0f;
    __syncthreads();

    const __bf16* arow = F   + (size_t)rowA * C2;
    const __bf16* brow = WpB + (size_t)ncol * C2;

    v8f acc = {};
    for (int kb = 0; kb < C2; kb += 32) {
        v8bf a0 = *(const v8bf*)(arow + kb + 8 * h);
        v8bf a1 = *(const v8bf*)(arow + kb + 16 + 8 * h);
        v8bf b0 = *(const v8bf*)(brow + kb + 8 * h);
        v8bf b1 = *(const v8bf*)(brow + kb + 16 + 8 * h);
        v16bf a, bb;
        #pragma unroll
        for (int e = 0; e < 8; ++e) { a[e] = a0[e]; a[8 + e] = a1[e]; bb[e] = b0[e]; bb[8 + e] = b1[e]; }
        acc = __builtin_amdgcn_wmma_f32_16x16x32_bf16(false, a, false, bb, (short)0, acc, false, false);
    }

    // cross-wave row sum of squares (C elem r belongs to row r+8h, col wave*16+m)
    #pragma unroll
    for (int r = 0; r < 8; ++r) atomicAdd(&rowNorm[r + 8 * h], acc[r] * acc[r]);
    __syncthreads();

    #pragma unroll
    for (int r = 0; r < 8; ++r) {
        float inv = 1.0f / fmaxf(sqrtf(rowNorm[r + 8 * h]), 1e-12f);
        int rr = mtile * 16 + r + 8 * h;
        Xn[(size_t)rr * D_DIM + wave * 16 + m] = f2bf(acc[r] * inv);
    }
}

// ---------------------------------------------------------------------------
// Kernel C: banded similarity (16x128 block per tile) + window-FC + bias + ReLU.
// sim block cols map to s = t0 - 50 + col (zero-masked outside batch).
// windows[i][w] = simblock[i][i + w]  (i+w <= 115 < 128).
// FC: K = 101 zero-padded to 128; W_fc pre-padded in bf16.
// ---------------------------------------------------------------------------
__global__ void sim_fc_kernel(const __bf16* __restrict__ Xn, const __bf16* __restrict__ WfcB,
                              const float* __restrict__ bfc, float* __restrict__ out) {
    __shared__ float simL[16 * 132];
    int wave = threadIdx.x >> 5;
    int lane = threadIdx.x & 31;
    int m = lane & 15, h = lane >> 4;
    int tile = blockIdx.x;                             // 128 tiles total
    int b  = tile >> 5;                                // 32 tiles per batch
    int t0 = (tile & 31) * 16;

    // ---- sim block: 8 waves each produce a 16x16 slice of the 16x128 block ----
    int colB = wave * 16 + m;                          // block column
    int s    = t0 - PAD + colB;                        // local s inside batch
    int okB  = (s >= 0 && s < T_DIM) ? 1 : 0;
    const __bf16* arow = Xn + (size_t)(b * T_DIM + t0 + m) * D_DIM;
    const __bf16* brow = Xn + (size_t)(b * T_DIM + (okB ? s : 0)) * D_DIM;

    v8f acc = {};
    #pragma unroll
    for (int kb = 0; kb < D_DIM; kb += 32) {
        v8bf a0 = *(const v8bf*)(arow + kb + 8 * h);
        v8bf a1 = *(const v8bf*)(arow + kb + 16 + 8 * h);
        v8bf b0 = *(const v8bf*)(brow + kb + 8 * h);
        v8bf b1 = *(const v8bf*)(brow + kb + 16 + 8 * h);
        __bf16 msk = okB ? f2bf(1.0f) : f2bf(0.0f);    // zero B for out-of-range s
        v16bf a, bb;
        #pragma unroll
        for (int e = 0; e < 8; ++e) {
            a[e] = a0[e]; a[8 + e] = a1[e];
            bb[e] = (__bf16)(b0[e] * msk); bb[8 + e] = (__bf16)(b1[e] * msk);
        }
        acc = __builtin_amdgcn_wmma_f32_16x16x32_bf16(false, a, false, bb, (short)0, acc, false, false);
    }

    #pragma unroll
    for (int r = 0; r < 8; ++r) simL[(r + 8 * h) * 132 + wave * 16 + m] = acc[r];
    __syncthreads();

    // ---- window FC: A = band-extracted windows (bf16), B = padded W_fc row ----
    int ocol = wave * 16 + m;                          // output feature
    const __bf16* wrow = WfcB + (size_t)ocol * 128;
    v8f acc2 = {};
    #pragma unroll
    for (int kb = 0; kb < 128; kb += 32) {
        v16bf a, bb;
        #pragma unroll
        for (int e = 0; e < 16; ++e) {
            int w = kb + ((e < 8) ? (8 * h + e) : (16 + 8 * h + (e - 8)));
            a[e] = (w < LWIN) ? f2bf(simL[m * 132 + (m + w)]) : f2bf(0.0f);
        }
        v8bf b0 = *(const v8bf*)(wrow + kb + 8 * h);
        v8bf b1 = *(const v8bf*)(wrow + kb + 16 + 8 * h);
        #pragma unroll
        for (int e = 0; e < 8; ++e) { bb[e] = b0[e]; bb[8 + e] = b1[e]; }
        acc2 = __builtin_amdgcn_wmma_f32_16x16x32_bf16(false, a, false, bb, (short)0, acc2, false, false);
    }

    float bias = bfc[ocol];
    #pragma unroll
    for (int r = 0; r < 8; ++r) {
        int t = t0 + r + 8 * h;
        float v = acc2[r] + bias;
        out[(size_t)(b * T_DIM + t) * D_DIM + ocol] = fmaxf(v, 0.0f);
    }
}

// ---------------------------------------------------------------------------
// Launch
// ---------------------------------------------------------------------------
extern "C" void kernel_launch(void* const* d_in, const int* in_sizes, int n_in,
                              void* d_out, int out_size, void* d_ws, size_t ws_size,
                              hipStream_t stream) {
    (void)in_sizes; (void)n_in; (void)out_size; (void)ws_size;
    const float* x1   = (const float*)d_in[0];
    const float* x2   = (const float*)d_in[1];
    const float* Wp   = (const float*)d_in[2];
    const float* Wfc  = (const float*)d_in[3];
    const float* bfc  = (const float*)d_in[4];
    float* out = (float*)d_out;

    char* ws = (char*)d_ws;
    __bf16* F    = (__bf16*)(ws);                                   // 2048*2048*2 = 8 MiB
    __bf16* Xnrm = (__bf16*)(ws + (size_t)2048 * 2048 * 2);         // 2048*128*2
    __bf16* WpB  = (__bf16*)(ws + (size_t)2048 * 2048 * 2 + 2048 * 128 * 2);
    __bf16* WfcB = (__bf16*)(ws + (size_t)2048 * 2048 * 2 + 2048 * 128 * 2 + 128 * 2048 * 2);

    prep_kernel  <<<(D_DIM * C2 + 255) / 256, 256, 0, stream>>>(Wp, Wfc, WpB, WfcB);
    mean_kernel  <<<(2048 * 2048) / 256,       256, 0, stream>>>(x1, x2, F);
    proj_kernel  <<<128,                       256, 0, stream>>>(F, WpB, Xnrm);
    sim_fc_kernel<<<128,                       256, 0, stream>>>(Xnrm, WfcB, bfc, out);
}
